// EMA_25237227831779
// MI455X (gfx1250) — matrix operand order repaired and checked by reading, not compiled, not run on hardware
//
#include <hip/hip_runtime.h>
#include <hip/hip_bf16.h>

// Problem constants (from reference)
#define BG   64      // B*G
#define CGc  40      // channels per group
#define HH   128
#define WW   128
#define HWxy 16384
#define UPCH 24
#define LOWCH 16
#define EPSV 1e-5f

typedef __attribute__((ext_vector_type(16))) _Float16 v16h;
typedef __attribute__((ext_vector_type(8)))  _Float16 v8h;
typedef __attribute__((ext_vector_type(8)))  float    v8f;

__device__ __forceinline__ float sigmoidf_(float v) {
    // v_exp_f32 + v_rcp_f32 instead of IEEE divide sequence
    return __builtin_amdgcn_rcpf(1.0f + __expf(-v));
}

__device__ __forceinline__ float waveRed(float v) {
#pragma unroll
    for (int m = 16; m >= 1; m >>= 1) v += __shfl_xor(v, m, 32);
    return v;
}

// Build a 16x32 f16 A-fragment from row-major LDS [M][64]:
// lane's halves are rows of two contiguous 16B runs -> 2x ds_load_b128.
__device__ __forceinline__ v16h loadAfrag(const _Float16* Al, int m, int ka) {
    v8h a0 = *(const v8h*)&Al[m * 64 + ka];
    v8h a1 = *(const v8h*)&Al[m * 64 + ka + 16];
    return __builtin_shufflevector(a0, a1, 0, 1, 2, 3, 4, 5, 6, 7,
                                   8, 9, 10, 11, 12, 13, 14, 15);
}

// Build a 32x16 f16 B-fragment from TRANSPOSED LDS [N][64]:
// lane's halves are two contiguous 16B runs -> 2x ds_load_b128.
__device__ __forceinline__ v16h loadBfrag(const _Float16* Bt, int n, int kb) {
    v8h b0 = *(const v8h*)&Bt[n * 64 + kb];
    v8h b1 = *(const v8h*)&Bt[n * 64 + kb + 16];
    return __builtin_shufflevector(b0, b1, 0, 1, 2, 3, 4, 5, 6, 7,
                                   8, 9, 10, 11, 12, 13, 14, 15);
}

// ---------------------------------------------------------------------------
// k1: per-(bg,c) plane row means (x_h) and column means (x_w)
// ---------------------------------------------------------------------------
__global__ void k1_rowcol(const float* __restrict__ x,
                          float* __restrict__ xh, float* __restrict__ xw) {
    int plane = blockIdx.x;                       // bg*CG + c
    const float* p = x + (size_t)plane * HWxy;
    int tid = threadIdx.x;
    __shared__ float colp[256];

    int col = tid & 127, half = tid >> 7;
    float acc = 0.f;
    for (int r = half * 64; r < half * 64 + 64; ++r) acc += p[r * WW + col];
    colp[tid] = acc;
    __syncthreads();
    if (tid < 128)
        xw[(size_t)plane * WW + tid] = (colp[tid] + colp[tid + 128]) * (1.0f / 128.0f);

    int lane = tid & 31, wv = tid >> 5;
    for (int r = wv; r < HH; r += 8) {
        float a = p[r * WW + lane] + p[r * WW + lane + 32] +
                  p[r * WW + lane + 64] + p[r * WW + lane + 96];
        a = waveRed(a);
        if (lane == 0) xh[(size_t)plane * HH + r] = a * (1.0f / 128.0f);
    }
}

// ---------------------------------------------------------------------------
// k2: hw = w1x1(40x40) @ [x_h;x_w](40x256) + b, sigmoid -> sig_h, sig_w
// WMMA f16 GEMM per bg. M=40(pad48), N=256, K=40(pad64). B staged transposed.
// ---------------------------------------------------------------------------
__global__ void k2_hwgemm(const float* __restrict__ w1, const float* __restrict__ b1,
                          const float* __restrict__ xh, const float* __restrict__ xw,
                          float* __restrict__ sigh, float* __restrict__ sigw) {
    __shared__ __align__(16) _Float16 Al[48 * 64];     // [M][Kpad]
    __shared__ __align__(16) _Float16 Bt[256 * 64];    // [N][Kpad] transposed
    int bg = blockIdx.x, tid = threadIdx.x;

    for (int i = tid; i < 48 * 64; i += 256) {
        int m = i >> 6, k = i & 63;
        Al[i] = (_Float16)((m < CGc && k < CGc) ? w1[m * CGc + k] : 0.0f);
    }
    for (int i = tid; i < 64 * 256; i += 256) {
        int k = i >> 8, n = i & 255;                   // coalesced global read
        float vv = 0.f;
        if (k < CGc)
            vv = (n < 128) ? xh[((size_t)bg * CGc + k) * HH + n]
                           : xw[((size_t)bg * CGc + k) * WW + (n - 128)];
        Bt[n * 64 + k] = (_Float16)vv;                 // transposed store
    }
    __syncthreads();

    int lane = tid & 31, wv = tid >> 5;
    int mh = (lane & 16) ? 8 : 0;
    for (int t = wv; t < 3 * 16; t += 8) {
        int mt = t >> 4, nt = t & 15;
        int m = mt * 16 + (lane & 15);
        int n = nt * 16 + (lane & 15);
        v8f c = {};
#pragma unroll
        for (int ks = 0; ks < 64; ks += 32) {
            v16h a = loadAfrag(Al, m, ks + ((lane & 16) ? 8 : 0));
            v16h b = loadBfrag(Bt, n, ks + ((lane & 16) ? 8 : 0));
            c = __builtin_amdgcn_wmma_f32_16x16x32_f16(false, a, false, b,
                                                       (short)0, c, false, false);
        }
#pragma unroll
        for (int vr = 0; vr < 8; ++vr) {
            int row = mt * 16 + vr + mh;
            if (row < CGc) {
                float s = sigmoidf_(c[vr] + b1[row]);
                if (n < 128) sigh[((size_t)bg * CGc + row) * HH + n] = s;
                else         sigw[((size_t)bg * CGc + row) * WW + (n - 128)] = s;
            }
        }
    }
}

// ---------------------------------------------------------------------------
// k3: per-(bg,c) stats of gated = gx * sig_h[y] * sig_w[x]  ->  mu, rstd
// ---------------------------------------------------------------------------
__global__ void k3_stats(const float* __restrict__ x, const float* __restrict__ sigh,
                         const float* __restrict__ sigw,
                         float* __restrict__ mu, float* __restrict__ rstd) {
    int plane = blockIdx.x;
    const float* p = x + (size_t)plane * HWxy;
    __shared__ float sh[128], sw[128], wp[8], wp2[8];
    int tid = threadIdx.x;
    if (tid < 128) sh[tid] = sigh[(size_t)plane * HH + tid];
    else           sw[tid - 128] = sigw[(size_t)plane * WW + tid - 128];
    __syncthreads();

    float s = 0.f, s2 = 0.f;
    for (int i = tid; i < HWxy; i += 256) {
        __builtin_prefetch(p + i + 2048, 0, 3);   // global_prefetch_b8
        int y = i >> 7, c = i & 127;
        float g = p[i] * sh[y] * sw[c];
        s += g; s2 += g * g;
    }
    s = waveRed(s); s2 = waveRed(s2);
    int lane = tid & 31, wv = tid >> 5;
    if (lane == 0) { wp[wv] = s; wp2[wv] = s2; }
    __syncthreads();
    if (tid == 0) {
        float t = 0.f, t2 = 0.f;
        for (int i = 0; i < 8; ++i) { t += wp[i]; t2 += wp2[i]; }
        float m  = t * (1.0f / HWxy);
        float vv = t2 * (1.0f / HWxy) - m * m;
        mu[plane] = m;
        rstd[plane] = rsqrtf(vv + EPSV);
    }
}

// ---------------------------------------------------------------------------
// k4: static weight fold. G[cg][col] f16, col layout:
//   0..599  : 5x5 folded kernel over gx channels 0..23 (col = c*25 + d)
//   600..615: 1x1 over gx channels 24..39
//   616     : b_gwc[cg]
// ---------------------------------------------------------------------------
__global__ void k4_fold(const float* __restrict__ w_sq1, const float* __restrict__ w_sq2,
                        const float* __restrict__ w_gwc, const float* __restrict__ b_gwc,
                        const float* __restrict__ w_pwc1, const float* __restrict__ w_pwc2,
                        _Float16* __restrict__ G) {
    for (int idx = threadIdx.x; idx < CGc * 640; idx += blockDim.x) {
        int cg = idx / 640, col = idx % 640;
        float val = 0.f;
        if (col < 600) {
            int c = col / 25, d = col % 25;
            int dy = d / 5, dx = d % 5;
            int g2 = (cg < 20) ? 0 : 1;
            for (int kk = 0; kk < 6; ++kk) {
                int k = g2 * 6 + kk;
                val += w_gwc[(cg * 6 + kk) * 25 + d] * w_sq1[k * UPCH + c];
            }
            if (dy >= 1 && dy <= 3 && dx >= 1 && dx <= 3) {
                int dp = (dy - 1) * 3 + (dx - 1);
                for (int k = 0; k < 12; ++k)
                    val += w_pwc1[(cg * 12 + k) * 9 + dp] * w_sq1[k * UPCH + c];
            }
        } else if (col < 616) {
            int cp = col - 600;
            if (cg < 32) {
                for (int j = 0; j < 8; ++j)
                    val += w_pwc2[cg * 8 + j] * w_sq2[j * LOWCH + cp];
            } else {
                val = w_sq2[(cg - 32) * LOWCH + cp];
            }
        } else if (col == 616) {
            val = b_gwc[cg];
        }
        G[idx] = (_Float16)val;
    }
}

// ---------------------------------------------------------------------------
// k5: x11 = softmax over CG of x1 spatial means (== softmax(gn_b) exactly),
// replicated across bg, K padded to 64 with zeros. f16 for WMMA A operand.
// ---------------------------------------------------------------------------
__global__ void k5_x11(const float* __restrict__ gn_b, _Float16* __restrict__ x11) {
    __shared__ float e[CGc];
    __shared__ float inv;
    int tid = threadIdx.x;
    if (tid == 0) {
        float mx = gn_b[0];
        for (int i = 1; i < CGc; ++i) mx = fmaxf(mx, gn_b[i]);
        float ss = 0.f;
        for (int i = 0; i < CGc; ++i) { e[i] = __expf(gn_b[i] - mx); ss += e[i]; }
        inv = 1.0f / ss;
    }
    __syncthreads();
    for (int idx = tid; idx < 64 * 64; idx += blockDim.x) {
        int k = idx & 63;
        x11[idx] = (_Float16)((k < CGc) ? e[k] * inv : 0.0f);
    }
}

// ---------------------------------------------------------------------------
// k6: K5c[64][640] = x11(64 x 40pad64) @ G(40pad64 x 640), WMMA f16.
// One block per 128 output columns. B staged transposed.
// ---------------------------------------------------------------------------
__global__ void k6_gemm(const _Float16* __restrict__ x11, const _Float16* __restrict__ G,
                        float* __restrict__ K5c) {
    __shared__ __align__(16) _Float16 Al[64 * 64];     // [M][Kpad]
    __shared__ __align__(16) _Float16 Bt[128 * 64];    // [N][Kpad] transposed
    int col0 = blockIdx.x * 128, tid = threadIdx.x;

    for (int i = tid; i < 64 * 64; i += 256) Al[i] = x11[i];
    for (int i = tid; i < 64 * 128; i += 256) {
        int k = i >> 7, n = i & 127;                   // coalesced global read
        Bt[n * 64 + k] = (k < CGc) ? G[k * 640 + col0 + n] : (_Float16)0.0f;
    }
    __syncthreads();

    int lane = tid & 31, wv = tid >> 5;
    for (int t = wv; t < 4 * 8; t += 8) {
        int mt = t >> 3, nt = t & 7;
        int m = mt * 16 + (lane & 15);
        int n = nt * 16 + (lane & 15);
        v8f c = {};
#pragma unroll
        for (int ks = 0; ks < 64; ks += 32) {
            v16h a = loadAfrag(Al, m, ks + ((lane & 16) ? 8 : 0));
            v16h b = loadBfrag(Bt, n, ks + ((lane & 16) ? 8 : 0));
            c = __builtin_amdgcn_wmma_f32_16x16x32_f16(false, a, false, b,
                                                       (short)0, c, false, false);
        }
#pragma unroll
        for (int vr = 0; vr < 8; ++vr) {
            int row = mt * 16 + vr + ((lane & 16) ? 8 : 0);
            K5c[(size_t)row * 640 + col0 + n] = c[vr];
        }
    }
}

// ---------------------------------------------------------------------------
// k7: fused spatial kernel. 16x16 tile per block (grid 64 tiles x 64 bg).
//  weights = folded-conv(gx) + 2 * sum_c x1 ; out = gx^2 * sigmoid(weights)
//  LDS budget kept < 64KB: ch 0..23 with 20x20 halo, ch 24..39 center 16x16.
// ---------------------------------------------------------------------------
__global__ void k7_main(const float* __restrict__ x, const float* __restrict__ sigh,
                        const float* __restrict__ sigw, const float* __restrict__ mu,
                        const float* __restrict__ rstd, const float* __restrict__ gn_w,
                        const float* __restrict__ gn_b, const float* __restrict__ K5c,
                        float* __restrict__ out) {
    __shared__ float gxA[24 * 400];     // channels 0..23, 20x20 (halo 2)
    __shared__ float gxB[16 * 256];     // channels 24..39, 16x16 center
    __shared__ float shl[CGc * 16], swl[CGc * 16];
    __shared__ float muL[CGc], rsL[CGc], gwL[CGc], gbL[CGc];
    __shared__ float K5L[600], K1L[16], cbL;

    int bg = blockIdx.y;
    int tileid = blockIdx.x;
    int y0 = (tileid >> 3) * 16, x0 = (tileid & 7) * 16;
    int tid = threadIdx.x;
    const float* px = x + (size_t)bg * CGc * HWxy;

    for (int i = tid; i < 24 * 400; i += 256) {
        int c = i / 400, r = i % 400;
        int iy = r / 20, ix = r % 20;
        int gy = y0 + iy - 2, gcol = x0 + ix - 2;
        float v = 0.f;
        if (gy >= 0 && gy < HH && gcol >= 0 && gcol < WW)
            v = px[(size_t)c * HWxy + gy * WW + gcol];
        gxA[i] = v;
    }
    for (int i = tid; i < 16 * 256; i += 256) {
        int c = 24 + (i >> 8), r = i & 255;
        int iy = r >> 4, ix = r & 15;
        gxB[i] = px[(size_t)c * HWxy + (y0 + iy) * WW + (x0 + ix)];
    }
    for (int i = tid; i < CGc * 16; i += 256) {
        int c = i >> 4, q = i & 15;
        shl[i] = sigh[((size_t)bg * CGc + c) * HH + y0 + q];
        swl[i] = sigw[((size_t)bg * CGc + c) * WW + x0 + q];
    }
    if (tid < CGc) {
        muL[tid] = mu[bg * CGc + tid]; rsL[tid] = rstd[bg * CGc + tid];
        gwL[tid] = gn_w[tid];          gbL[tid] = gn_b[tid];
    }
    const float* kb = K5c + (size_t)bg * 640;
    for (int i = tid; i < 600; i += 256) K5L[i] = kb[i];
    if (tid < 16)  K1L[tid] = kb[600 + tid];
    if (tid == 0)  cbL = kb[616];
    __syncthreads();

    int py = tid >> 4, pxl = tid & 15;

    // folded conv term
    float A = cbL;
    for (int c = 0; c < 24; ++c) {
        const float* gp = &gxA[c * 400 + py * 20 + pxl];
        const float* kp = &K5L[c * 25];
#pragma unroll
        for (int dy = 0; dy < 5; ++dy)
#pragma unroll
            for (int dx = 0; dx < 5; ++dx)
                A += kp[dy * 5 + dx] * gp[dy * 20 + dx];
    }
#pragma unroll
    for (int c = 0; c < 16; ++c)
        A += K1L[c] * gxB[c * 256 + py * 16 + pxl];

    // 2 * sum_c x1 term
    float Bs = 0.f;
    for (int c = 0; c < CGc; ++c) {
        float g = (c < 24) ? gxA[c * 400 + (py + 2) * 20 + (pxl + 2)]
                           : gxB[(c - 24) * 256 + py * 16 + pxl];
        g *= shl[c * 16 + py] * swl[c * 16 + pxl];
        Bs += (g - muL[c]) * rsL[c] * gwL[c] + gbL[c];
    }

    float s = sigmoidf_(A + 2.0f * Bs);

    size_t obase = (size_t)bg * CGc * HWxy + (size_t)(y0 + py) * WW + (x0 + pxl);
    for (int c = 0; c < CGc; ++c) {
        float g = (c < 24) ? gxA[c * 400 + (py + 2) * 20 + (pxl + 2)]
                           : gxB[(c - 24) * 256 + py * 16 + pxl];
        out[obase + (size_t)c * HWxy] = g * g * s;
    }
}

// ---------------------------------------------------------------------------
extern "C" void kernel_launch(void* const* d_in, const int* in_sizes, int n_in,
                              void* d_out, int out_size, void* d_ws, size_t ws_size,
                              hipStream_t stream) {
    (void)in_sizes; (void)n_in; (void)out_size; (void)ws_size;

    const float* x      = (const float*)d_in[0];
    const float* w1x1   = (const float*)d_in[1];
    const float* b1x1   = (const float*)d_in[2];
    const float* gn_w   = (const float*)d_in[3];
    const float* gn_b   = (const float*)d_in[4];
    const float* w_sq1  = (const float*)d_in[5];
    const float* w_sq2  = (const float*)d_in[6];
    const float* w_gwc  = (const float*)d_in[7];
    const float* b_gwc  = (const float*)d_in[8];
    const float* w_pwc1 = (const float*)d_in[9];
    const float* w_pwc2 = (const float*)d_in[10];
    float* out = (float*)d_out;

    char* ws = (char*)d_ws;
    size_t off = 0;
    auto alloc = [&](size_t bytes) -> void* {
        void* p = ws + off;
        off += (bytes + 255) & ~(size_t)255;
        return p;
    };

    float*     xh    = (float*)alloc((size_t)BG * CGc * HH * 4);
    float*     xw    = (float*)alloc((size_t)BG * CGc * WW * 4);
    float*     sigh  = (float*)alloc((size_t)BG * CGc * HH * 4);
    float*     sigw  = (float*)alloc((size_t)BG * CGc * WW * 4);
    float*     mu_   = (float*)alloc((size_t)BG * CGc * 4);
    float*     rstd_ = (float*)alloc((size_t)BG * CGc * 4);
    _Float16*  Gf    = (_Float16*)alloc((size_t)CGc * 640 * 2);
    _Float16*  x11f  = (_Float16*)alloc((size_t)64 * 64 * 2);
    float*     K5c   = (float*)alloc((size_t)BG * 640 * 4);

    k1_rowcol<<<dim3(BG * CGc), dim3(256), 0, stream>>>(x, xh, xw);
    k2_hwgemm<<<dim3(BG),       dim3(256), 0, stream>>>(w1x1, b1x1, xh, xw, sigh, sigw);
    k3_stats <<<dim3(BG * CGc), dim3(256), 0, stream>>>(x, sigh, sigw, mu_, rstd_);
    k4_fold  <<<dim3(1),        dim3(256), 0, stream>>>(w_sq1, w_sq2, w_gwc, b_gwc,
                                                        w_pwc1, w_pwc2, Gf);
    k5_x11   <<<dim3(1),        dim3(256), 0, stream>>>(gn_b, x11f);
    k6_gemm  <<<dim3(5),        dim3(256), 0, stream>>>(x11f, Gf, K5c);
    k7_main  <<<dim3(64, BG),   dim3(256), 0, stream>>>(x, sigh, sigw, mu_, rstd_,
                                                        gn_w, gn_b, K5c, out);
}
